// ConvolutionalMultiheadAttention_5368709120308
// MI455X (gfx1250) — compile-verified, matrix operand order checked by reading, not run on hardware
//
#include <hip/hip_runtime.h>

// ---------------------------------------------------------------------------
// MI455X (gfx1250): conv1d(x3) as f16 WMMA GEMMs + flash-attention.
// Compute-bound (~111 GFLOP vs ~0.14 GB @ 23.3 TB/s) -> f16 WMMA, f32 accum.
// Double-buffered LDS tiles fed by GLOBAL_LOAD_ASYNC_TO_LDS_B128 (ASYNCcnt),
// so the tile t+1 copy overlaps tile t's WMMAs; explicit split-barrier +
// counter waits instead of __syncthreads to preserve the overlap.
// ---------------------------------------------------------------------------

typedef __attribute__((ext_vector_type(16))) _Float16 v16h;
typedef __attribute__((ext_vector_type(8)))  float    v8f;
typedef __attribute__((ext_vector_type(4)))  int      v4i;

#define C_DIM 512
#define L_DIM 1024
#define LP    1022
#define BATCH 16
#define HEADS 8
#define DH    64

#if defined(__HIP_DEVICE_COMPILE__) && __has_builtin(__builtin_amdgcn_global_load_async_to_lds_b128)
#define ASYNC_LDS 1
#else
#define ASYNC_LDS 0
#endif

#define WAIT_ASYNC0() asm volatile("s_wait_asynccnt 0x0" ::: "memory")
#define WAIT_DS0()    asm volatile("s_wait_dscnt 0x0" ::: "memory")
#define WG_BARRIER()  asm volatile("s_barrier_signal -1\n\ts_barrier_wait -1" ::: "memory")

#if ASYNC_LDS
__device__ __forceinline__ void copy16_async(_Float16* l, const _Float16* g) {
  __builtin_amdgcn_global_load_async_to_lds_b128(
      (__attribute__((address_space(1))) v4i*)const_cast<_Float16*>(g),
      (__attribute__((address_space(3))) v4i*)l, 0, 0);
}
#endif

struct HalfFrag { uint4 lo, hi; };

// 16x32 f16 A/B fragment (CDNA5 16-bit layout) from row-major (row,k) buffer:
// lane row = lane&15, kbase = (lane&16)?8:0; halves [kb..kb+7],[kb+16..kb+23]
// -> two b128 loads.  All strides used are multiples of 8 halves (16B).
__device__ __forceinline__ v16h load_frag(const _Float16* base, int stride, int lane) {
  const int m  = lane & 15;
  const int kb = (lane & 16) ? 8 : 0;
  const _Float16* p = base + m * stride + kb;
  HalfFrag t;
  t.lo = *(const uint4*)(p);
  t.hi = *(const uint4*)(p + 16);
  return __builtin_bit_cast(v16h, t);
}

__device__ __forceinline__ v8f wmma16(v16h a, v16h b, v8f c) {
  return __builtin_amdgcn_wmma_f32_16x16x32_f16(false, a, false, b, (short)0, c,
                                                false, false);
}

// ---------------------------------------------------------------------------
// Pack kernels: w -> [conv][shift][O][I] f16 ; x -> x^T [B][1026][C] f16.
// ---------------------------------------------------------------------------
__global__ void pack_w_kernel(const float* __restrict__ w0,
                              const float* __restrict__ w1,
                              const float* __restrict__ w2,
                              _Float16* __restrict__ wPack) {
  int t = blockIdx.x * 256 + threadIdx.x;
  if (t >= 3 * C_DIM * C_DIM) return;
  int conv = t / (C_DIM * C_DIM);
  int rem  = t % (C_DIM * C_DIM);
  int o = rem / C_DIM, i = rem % C_DIM;
  const float* w = (conv == 0) ? w0 : (conv == 1) ? w1 : w2;
#pragma unroll
  for (int s = 0; s < 3; ++s)
    wPack[(((size_t)conv * 3 + s) * C_DIM + o) * C_DIM + i] =
        (_Float16)w[((size_t)o * C_DIM + i) * 3 + s];
}

__global__ void pack_x_kernel(const float* __restrict__ x, _Float16* __restrict__ xT) {
  size_t t = (size_t)blockIdx.x * 256 + threadIdx.x;
  if (t >= (size_t)BATCH * 1026 * C_DIM) return;
  int c = (int)(t % C_DIM);
  int l = (int)((t / C_DIM) % 1026);
  int b = (int)(t / ((size_t)C_DIM * 1026));
  float v = (l < L_DIM) ? x[((size_t)b * C_DIM + c) * L_DIM + l] : 0.f;
  xT[t] = (_Float16)v;
}

// ---------------------------------------------------------------------------
// Conv1d as GEMM with double-buffered async tiles.
// Block 256 thr = 8 waves (2x4); tile 128(M=o) x 128(N=l); wave 64x32;
// K-loop: 48 chunks (3 shifts x 16 x K32).  8 WMMA / wave / chunk.
// ---------------------------------------------------------------------------
__global__ __launch_bounds__(256) void conv_gemm_kernel(
    const _Float16* __restrict__ wPack, const _Float16* __restrict__ xT,
    const float* __restrict__ b0, const float* __restrict__ b1,
    const float* __restrict__ b2,
    _Float16* __restrict__ Qp, _Float16* __restrict__ Kp,
    _Float16* __restrict__ Vp) {
  __shared__ _Float16 sA[2][128 * 40];   // stride 40 halves = 80B (16B mult)
  __shared__ _Float16 sB[2][128 * 40];

  const int tid  = threadIdx.x;
  const int lane = tid & 31;
  const int wave = tid >> 5;
  const int conv = blockIdx.z;
  const int b    = blockIdx.y;
  const int mt   = (blockIdx.x & 3) * 128;
  const int nt   = (blockIdx.x >> 2) * 128;
  const int wm   = (wave >> 2) * 64;
  const int wn   = (wave & 3) * 32;

  v8f acc[4][2] = {};

  auto copy_tile = [&](int t, int pb) {
    const int s = t >> 4, k0 = (t & 15) << 5;
    const _Float16* Ab = wPack + (((size_t)conv * 3 + s) * C_DIM + mt) * C_DIM + k0;
    const _Float16* Bb = xT + ((size_t)b * 1026 + nt + s) * C_DIM + k0;
#if ASYNC_LDS
#pragma unroll
    for (int u0 = 0; u0 < 2; ++u0) {
      int u = tid + u0 * 256, row = u >> 2, c8 = (u & 3) * 8;
      copy16_async(&sA[pb][row * 40 + c8], Ab + (size_t)row * C_DIM + c8);
      copy16_async(&sB[pb][row * 40 + c8], Bb + (size_t)row * C_DIM + c8);
    }
#else
    uint4 ra[2], rb[2];
#pragma unroll
    for (int u0 = 0; u0 < 2; ++u0) {
      int u = tid + u0 * 256, row = u >> 2, c8 = (u & 3) * 8;
      ra[u0] = *(const uint4*)(Ab + (size_t)row * C_DIM + c8);
      rb[u0] = *(const uint4*)(Bb + (size_t)row * C_DIM + c8);
    }
#pragma unroll
    for (int u0 = 0; u0 < 2; ++u0) {
      int u = tid + u0 * 256, row = u >> 2, c8 = (u & 3) * 8;
      *(uint4*)&sA[pb][row * 40 + c8] = ra[u0];
      *(uint4*)&sB[pb][row * 40 + c8] = rb[u0];
    }
#endif
  };

  copy_tile(0, 0);
  for (int t = 0; t < 48; ++t) {
    const int cur = t & 1;
    WAIT_ASYNC0();            // tile t resident in sA/sB[cur]
    WAIT_DS0();               // fallback producer stores visible
    WG_BARRIER();
    if (t + 1 < 48) copy_tile(t + 1, cur ^ 1);  // overlaps WMMAs below

    v16h af[4], bf[2];
#pragma unroll
    for (int i = 0; i < 4; ++i) af[i] = load_frag(&sA[cur][(wm + i * 16) * 40], 40, lane);
#pragma unroll
    for (int j = 0; j < 2; ++j) bf[j] = load_frag(&sB[cur][(wn + j * 16) * 40], 40, lane);
#pragma unroll
    for (int i = 0; i < 4; ++i)
#pragma unroll
      for (int j = 0; j < 2; ++j) acc[i][j] = wmma16(af[i], bf[j], acc[i][j]);
  }

  const float* bias = (conv == 0) ? b0 : (conv == 1) ? b1 : b2;
  const int nlane = lane & 15;
  const int hi    = (lane & 16) ? 8 : 0;
#pragma unroll
  for (int i = 0; i < 4; ++i)
#pragma unroll
    for (int j = 0; j < 2; ++j)
#pragma unroll
      for (int r = 0; r < 8; ++r) {
        int o = mt + wm + i * 16 + r + hi;   // C-frag: M = r (+8 upper half)
        int l = nt + wn + j * 16 + nlane;    //         N = lane&15
        float v = acc[i][j][r] + bias[o];
        if (conv == 2) {
          Vp[((size_t)b * C_DIM + o) * L_DIM + l] = (_Float16)v;
        } else {
          int h = o >> 6, d = o & 63;
          _Float16* T = (conv == 0) ? Qp : Kp;
          T[(((size_t)b * HEADS + h) * L_DIM + l) * DH + d] = (_Float16)v;
        }
      }
}

// ---------------------------------------------------------------------------
// Flash attention, double-buffered K/V tiles (64 kpos wide).
// Block 128 thr = 4 waves x 16 q rows.  Per k-block: 8 WMMA (S) + 8 WMMA (PV).
// ---------------------------------------------------------------------------
__global__ __launch_bounds__(128) void attn_kernel(
    const _Float16* __restrict__ Qp, const _Float16* __restrict__ Kp,
    const _Float16* __restrict__ Vp, float* __restrict__ out) {
  __shared__ _Float16 sK[2][64 * 72];     // [kpos][d]
  __shared__ _Float16 sV[2][64 * 72];     // [d][kpos]
  __shared__ _Float16 sP[4 * 16 * 72];    // per-wave [q][kpos]

  const int tid  = threadIdx.x;
  const int lane = tid & 31;
  const int wave = tid >> 5;
  const int h    = blockIdx.y;
  const int b    = blockIdx.z;
  const int q0   = blockIdx.x * 64 + wave * 16;
  const int nlane = lane & 15;
  const int hi    = (lane & 16) ? 8 : 0;
  const float scale = 0.044194173824159216f;  // 1/sqrt(512)

  const _Float16* qbase = Qp + (((size_t)b * HEADS + h) * L_DIM + q0) * DH;
  v16h qf[2];
  qf[0] = load_frag(qbase, DH, lane);
  qf[1] = load_frag(qbase + 32, DH, lane);

  float mrow[8], lrow[8];
  v8f o[4] = {};
#pragma unroll
  for (int r = 0; r < 8; ++r) { mrow[r] = -1e30f; lrow[r] = 0.f; }

  _Float16* myP = &sP[wave * 16 * 72];

  auto copy_tile = [&](int kb, int pb) {
    const int l0 = kb * 64;
    const _Float16* gK = Kp + (((size_t)b * HEADS + h) * L_DIM + l0) * DH;
    const _Float16* gV = Vp + ((size_t)b * C_DIM + h * DH) * L_DIM + l0;
#if ASYNC_LDS
#pragma unroll
    for (int u0 = 0; u0 < 4; ++u0) {
      int u = tid + u0 * 128, row = u >> 3, c8 = (u & 7) * 8;
      copy16_async(&sK[pb][row * 72 + c8], gK + (size_t)row * DH + c8);
      copy16_async(&sV[pb][row * 72 + c8], gV + (size_t)row * L_DIM + c8);
    }
#else
    uint4 rk[4], rv[4];
#pragma unroll
    for (int u0 = 0; u0 < 4; ++u0) {
      int u = tid + u0 * 128, row = u >> 3, c8 = (u & 7) * 8;
      rk[u0] = *(const uint4*)(gK + (size_t)row * DH + c8);
      rv[u0] = *(const uint4*)(gV + (size_t)row * L_DIM + c8);
    }
#pragma unroll
    for (int u0 = 0; u0 < 4; ++u0) {
      int u = tid + u0 * 128, row = u >> 3, c8 = (u & 7) * 8;
      *(uint4*)&sK[pb][row * 72 + c8] = rk[u0];
      *(uint4*)&sV[pb][row * 72 + c8] = rv[u0];
    }
#endif
  };

  copy_tile(0, 0);
  for (int kb = 0; kb < 16; ++kb) {
    const int cur = kb & 1;
    const int l0  = kb * 64;
    WAIT_ASYNC0();
    WAIT_DS0();
    WG_BARRIER();
    if (kb + 1 < 16) copy_tile(kb + 1, cur ^ 1);  // overlaps compute

    v8f s[4] = {};
#pragma unroll
    for (int nf = 0; nf < 4; ++nf)
#pragma unroll
      for (int c = 0; c < 2; ++c)
        s[nf] = wmma16(qf[c], load_frag(&sK[cur][(nf * 16) * 72 + c * 32], 72, lane),
                       s[nf]);

    // scale + mask kpos >= 1022
#pragma unroll
    for (int nf = 0; nf < 4; ++nf) {
      bool valid = (l0 + nf * 16 + nlane) < LP;
#pragma unroll
      for (int r = 0; r < 8; ++r) s[nf][r] = valid ? s[nf][r] * scale : -1e30f;
    }

    // online softmax: stats align with C-frag rows (vgpr r <-> row r / r+8)
    float alpha[8];
#pragma unroll
    for (int r = 0; r < 8; ++r) {
      float mx = fmaxf(fmaxf(s[0][r], s[1][r]), fmaxf(s[2][r], s[3][r]));
#pragma unroll
      for (int off = 8; off >= 1; off >>= 1) mx = fmaxf(mx, __shfl_xor(mx, off, 32));
      float mnew = fmaxf(mrow[r], mx);
      alpha[r] = __expf(mrow[r] - mnew);
      mrow[r] = mnew;
      float rs = 0.f;
#pragma unroll
      for (int nf = 0; nf < 4; ++nf) {
        float p = __expf(s[nf][r] - mnew);
        s[nf][r] = p;
        rs += p;
      }
#pragma unroll
      for (int off = 8; off >= 1; off >>= 1) rs += __shfl_xor(rs, off, 32);
      lrow[r] = lrow[r] * alpha[r] + rs;
    }

    // rescale O; spill P (f16) to wave-private LDS (q,kpos) row-major
#pragma unroll
    for (int nf = 0; nf < 4; ++nf)
#pragma unroll
      for (int r = 0; r < 8; ++r) {
        o[nf][r] *= alpha[r];
        myP[(r + hi) * 72 + nf * 16 + nlane] = (_Float16)s[nf][r];
      }
    WAIT_DS0();  // wave-local P store -> A-frag reload ordering

    // O += P * V^T
#pragma unroll
    for (int c = 0; c < 2; ++c) {
      v16h pf = load_frag(myP + c * 32, 72, lane);
#pragma unroll
      for (int nf = 0; nf < 4; ++nf)
        o[nf] = wmma16(pf, load_frag(&sV[cur][(nf * 16) * 72 + c * 32], 72, lane),
                       o[nf]);
    }
  }

  // out[b][h*64+d][q] = O[q][d] / l[q]
#pragma unroll
  for (int nf = 0; nf < 4; ++nf) {
    int d = nf * 16 + nlane;
#pragma unroll
    for (int r = 0; r < 8; ++r) {
      int q = q0 + r + hi;
      if (q < LP)
        out[((size_t)b * C_DIM + h * DH + d) * LP + q] = o[nf][r] / lrow[r];
    }
  }
}

// ---------------------------------------------------------------------------
// Host launcher.  Workspace (f16): xT[16][1026][512] | wPack[3][3][512][512]
// | Qp[16][8][1024][64] | Kp same | Vp[16][512][1024]  (~71.9 MB total)
// ---------------------------------------------------------------------------
extern "C" void kernel_launch(void* const* d_in, const int* in_sizes, int n_in,
                              void* d_out, int out_size, void* d_ws, size_t ws_size,
                              hipStream_t stream) {
  const float* x  = (const float*)d_in[0];
  const float* w0 = (const float*)d_in[1];
  const float* b0 = (const float*)d_in[2];
  const float* w1 = (const float*)d_in[3];
  const float* b1 = (const float*)d_in[4];
  const float* w2 = (const float*)d_in[5];
  const float* b2 = (const float*)d_in[6];
  float* out = (float*)d_out;

  char* ws = (char*)d_ws;
  _Float16* xT    = (_Float16*)(ws);
  _Float16* wPack = (_Float16*)(ws + 16809984);
  _Float16* Qp    = (_Float16*)(ws + 16809984 + 4718592);
  _Float16* Kp    = Qp + (size_t)BATCH * HEADS * L_DIM * DH;
  _Float16* Vp    = Kp + (size_t)BATCH * HEADS * L_DIM * DH;

  pack_w_kernel<<<(3 * C_DIM * C_DIM) / 256, 256, 0, stream>>>(w0, w1, w2, wPack);
  pack_x_kernel<<<(BATCH * 1026 * C_DIM) / 256, 256, 0, stream>>>(x, xT);
  conv_gemm_kernel<<<dim3(32, BATCH, 3), 256, 0, stream>>>(wPack, xT, b0, b1, b2,
                                                           Qp, Kp, Vp);
  attn_kernel<<<dim3(16, HEADS, BATCH), 128, 0, stream>>>(Qp, Kp, Vp, out);
}
